// LearnableLeeFilter_52158082842933
// MI455X (gfx1250) — compile-verified, matrix operand order checked by reading, not run on hardware
//
#include <hip/hip_runtime.h>
#include <math.h>

typedef __attribute__((ext_vector_type(16))) _Float16 v16h;
typedef __attribute__((ext_vector_type(8)))  float    v8f;

#define WMMA_F16(A, B, C) \
    __builtin_amdgcn_wmma_f32_16x16x32_f16(false, (A), false, (B), (short)0, (C), false, false)

__device__ __forceinline__ int reflect_i(int i, int n) {
    i = (i < 0) ? -i : i;
    i = (i >= n) ? (2 * n - 2 - i) : i;
    return i;
}

// Branch-free exact-GELU: gelu(v) = 0.5*v*(1+erf(v/sqrt(2))).
// erf via Abramowitz-Stegun 7.1.26 (max abs err 1.5e-7), fully straight-line:
// one v_rcp_f32, one v_exp_f32, ~10 FMAs. No exec-mask divergence.
__device__ __forceinline__ float gelu_exact(float v) {
    const float x  = v * 0.70710678118654752f;
    const float ax = fabsf(x);
    const float t  = __builtin_amdgcn_rcpf(fmaf(0.3275911f, ax, 1.0f));
    float p = fmaf(1.061405429f, t, -1.453152027f);
    p = fmaf(p, t, 1.421413741f);
    p = fmaf(p, t, -0.284496736f);
    p = fmaf(p, t, 0.254829592f);
    p = p * t;
    const float e = __expf(-ax * ax);
    float erfv = fmaf(-p, e, 1.0f);     // erf(|x|)
    erfv = copysignf(erfv, x);
    return 0.5f * v * (1.0f + erfv);
}

constexpr int CH    = 4;     // channels
constexpr int HIMG  = 512;
constexpr int WIMG  = 512;
constexpr int TH    = 16;    // tile height
constexpr int TW    = 32;    // tile width
constexpr int XS_H  = TH + 6;  // 22 (3-halo each side)
constexpr int XS_W  = TW + 6;  // 38
constexpr int PITCH = 40;      // padded LDS row pitch (floats)

__global__ __launch_bounds__(256)
void lee_filter_mlp_kernel(const float* __restrict__ x,
                           const float* __restrict__ W1,
                           const float* __restrict__ b1,
                           const float* __restrict__ W2,
                           const float* __restrict__ b2,
                           float* __restrict__ out)
{
    __shared__ float     xs[CH][XS_H][PITCH];   // 14080 B
    __shared__ float     cs1[CH][TH][PITCH];    // 10240 B
    __shared__ float     cs2[CH][TH][PITCH];    // 10240 B
    __shared__ _Float16  featL[8][16][16];      //  4096 B
    __shared__ _Float16  hL[8][16][32];         //  8192 B
    __shared__ float     outL[8][16][4];        //  2048 B

    const int tid  = threadIdx.x;
    const int bimg = blockIdx.z;
    const int ty0  = blockIdx.y * TH;
    const int tx0  = blockIdx.x * TW;

    const int lane = tid & 31;
    const int wave = tid >> 5;
    const int nn   = lane & 15;   // N in B/C/D fragments; M row in A fragments
    const int hi   = lane >> 4;   // half-wave select

    // ---- per-lane constant B fragments (W1^T two N-halves, W2^T) ----
    // 16-bit fragment layout (wave32): element t of a lane holds
    //   K = t + (t<8 ? (hi?8:0) : (hi?16:8))
    v16h bf0, bf1, bf2;
#pragma unroll
    for (int t = 0; t < 16; ++t) {
        const int K = t + ((t < 8) ? (hi ? 8 : 0) : (hi ? 16 : 8));
        bf0[t] = (K < 16) ? (_Float16)W1[nn * 16 + K]        : (_Float16)0.0f;
        bf1[t] = (K < 16) ? (_Float16)W1[(nn + 16) * 16 + K] : (_Float16)0.0f;
        bf2[t] = (nn < 4) ? (_Float16)W2[nn * 32 + K]        : (_Float16)0.0f;
    }

    // biases folded into the WMMA C operand: C/D element (lane,v) is
    // (M=v+8*hi, N=nn) and bias depends only on N -> broadcast across v.
    const float bias0 = b1[nn];
    const float bias1 = b1[nn + 16];
    const float bias2 = (nn < 4) ? b2[nn] : 0.0f;
    v8f cb0, cb1, cb2;
#pragma unroll
    for (int v = 0; v < 8; ++v) { cb0[v] = bias0; cb1[v] = bias1; cb2[v] = bias2; }

    // ---- stage 1: load reflect-padded tile into LDS ----
    for (int i = tid; i < CH * XS_H * XS_W; i += 256) {
        const int c  = i / (XS_H * XS_W);
        const int r  = (i / XS_W) % XS_H;
        const int q  = i % XS_W;
        const int gy = reflect_i(ty0 + r - 3, HIMG);
        const int gx = reflect_i(tx0 + q - 3, WIMG);
        xs[c][r][q] = x[(((size_t)bimg * CH + c) * HIMG + gy) * WIMG + gx];
    }
    __syncthreads();

    // ---- stage 2: vertical 7-tap sums of x and x^2 ----
    for (int i = tid; i < CH * TH * XS_W; i += 256) {
        const int c  = i / (TH * XS_W);
        const int yy = (i / XS_W) % TH;
        const int q  = i % XS_W;
        float s = 0.0f, s2 = 0.0f;
#pragma unroll
        for (int d = 0; d < 7; ++d) {
            const float v = xs[c][yy + d][q];
            s  += v;
            s2 += v * v;
        }
        cs1[c][yy][q] = s;
        cs2[c][yy][q] = s2;
    }
    __syncthreads();

    // ---- stage 3: each wave runs 4 WMMA sub-tiles of 16 pixels ----
    for (int s = 0; s < 4; ++s) {
        // feat: lane handles pixel nn, channels {2*hi, 2*hi+1}
        {
            const int p  = wave * 64 + s * 16 + nn;
            const int ty = p >> 5, tx = p & 31;
#pragma unroll
            for (int cc = 0; cc < 2; ++cc) {
                const int c  = hi * 2 + cc;
                const float xv = xs[c][ty + 3][tx + 3];
                float ms = 0.0f, ms2 = 0.0f;
#pragma unroll
                for (int d = 0; d < 7; ++d) {
                    ms  += cs1[c][ty][tx + d];
                    ms2 += cs2[c][ty][tx + d];
                }
                const float mean = ms  * (1.0f / 49.0f);
                const float msq  = ms2 * (1.0f / 49.0f);
                float var = msq - mean * mean;
                var = (var < 0.0f) ? 0.0f : var;
                featL[wave][nn][c]      = (_Float16)xv;
                featL[wave][nn][4 + c]  = (_Float16)mean;
                featL[wave][nn][8 + c]  = (_Float16)var;
                featL[wave][nn][12 + c] = (_Float16)msq;
            }
        }

        // gather A fragment (real K = 16, padded to 32 with zeros)
        v16h a;
#pragma unroll
        for (int t = 0; t < 16; ++t) {
            const int K = t + ((t < 8) ? (hi ? 8 : 0) : (hi ? 16 : 8));
            a[t] = (K < 16) ? featL[wave][nn][K] : (_Float16)0.0f;
        }

        // layer 1: two 16x16x32 WMMAs, bias pre-loaded in C
        const v8f acc0 = WMMA_F16(a, bf0, cb0);
        const v8f acc1 = WMMA_F16(a, bf1, cb1);

        // branch-free exact GELU, repack to A-layout via LDS
#pragma unroll
        for (int v = 0; v < 8; ++v) {
            const int mo = v + 8 * hi;
            hL[wave][mo][nn]      = (_Float16)gelu_exact(acc0[v]);
            hL[wave][mo][nn + 16] = (_Float16)gelu_exact(acc1[v]);
        }

        v16h a2;
#pragma unroll
        for (int t = 0; t < 16; ++t) {
            const int K = t + ((t < 8) ? (hi ? 8 : 0) : (hi ? 16 : 8));
            a2[t] = hL[wave][nn][K];
        }

        // layer 2: residual[16px x 4] (N padded to 16), K = 32 exact, b2 in C
        const v8f acc2 = WMMA_F16(a2, bf2, cb2);

        if (nn < 4) {
#pragma unroll
            for (int v = 0; v < 8; ++v) {
                const int mo = v + 8 * hi;
                const int p  = wave * 64 + s * 16 + mo;
                const int ty = p >> 5, tx = p & 31;
                outL[wave][mo][nn] = acc2[v] + xs[nn][ty + 3][tx + 3];
            }
        }

        // coalesced stores: half-wave writes 16 consecutive pixels of one channel
#pragma unroll
        for (int j = 0; j < 2; ++j) {
            const int c  = j * 2 + hi;
            const int p  = wave * 64 + s * 16 + nn;
            const int ty = p >> 5, tx = p & 31;
            const int gy = ty0 + ty, gx = tx0 + tx;
            out[(((size_t)bimg * CH + c) * HIMG + gy) * WIMG + gx] = outL[wave][nn][c];
        }
    }
}

extern "C" void kernel_launch(void* const* d_in, const int* in_sizes, int n_in,
                              void* d_out, int out_size, void* d_ws, size_t ws_size,
                              hipStream_t stream) {
    const float* x  = (const float*)d_in[0];
    const float* W1 = (const float*)d_in[1];
    const float* b1 = (const float*)d_in[2];
    const float* W2 = (const float*)d_in[3];
    const float* b2 = (const float*)d_in[4];
    float* out = (float*)d_out;

    const int batch = in_sizes[0] / (CH * HIMG * WIMG);  // 16
    dim3 grid(WIMG / TW, HIMG / TH, batch);
    lee_filter_mlp_kernel<<<grid, 256, 0, stream>>>(x, W1, b1, W2, b2, out);
}
